// xLSTMEncoder_45921790328943
// MI455X (gfx1250) — matrix operand order repaired and checked by reading, not compiled, not run on hardware
//
#include <hip/hip_runtime.h>

#define SS 1024
#define BB 16
#define HH 128
#define DD 256
#define N4 512

typedef __bf16 v16bf __attribute__((ext_vector_type(16)));
typedef float  v8f   __attribute__((ext_vector_type(8)));
typedef float  v4f   __attribute__((ext_vector_type(4)));
typedef int    v4i   __attribute__((ext_vector_type(4)));

union FragAB { v4i i[2]; v16bf v; unsigned short u[16]; };
union Acc8   { v8f v; v4f h[2]; float f[8]; };

__device__ __forceinline__ unsigned short f2bf(float f) {
    unsigned int u = __float_as_uint(f);
    return (unsigned short)((u + 0x7FFFu + ((u >> 16) & 1u)) >> 16);
}
__device__ __forceinline__ float bf2f(unsigned short b) {
    return __uint_as_float(((unsigned int)b) << 16);
}
__device__ __forceinline__ float sigm(float x) { return 1.0f / (1.0f + __expf(-x)); }
__device__ __forceinline__ float fast_tanh(float x) {
    float e = __expf(-2.0f * fabsf(x));
    float r = (1.0f - e) / (1.0f + e);
    return copysignf(r, x);
}

// ---------------------------------------------------------------------------
// K0: pack weights to bf16 in WMMA B-fragment order.
// B-fragment (16x16x32 bf16): lane l holds column n = l%16, K = (l/16)*8 + (e&7) + (e>>3)*16.
// Fragment storage: frag[(tile_id*32 + lane)*16 + e]
// ---------------------------------------------------------------------------
__global__ void k0_pack(const float* __restrict__ Wx, const float* __restrict__ Wh,
                        const float* __restrict__ Wq, const float* __restrict__ Wk,
                        const float* __restrict__ Wv, const float* __restrict__ Wo,
                        unsigned short* __restrict__ WxP, unsigned short* __restrict__ WhP,
                        unsigned short* __restrict__ WfP)
{
    int idx = blockIdx.x * 256 + threadIdx.x;
    if (idx < 131072) {                       // Wx_s [256,512]: 32 ntiles x 8 ksteps
        int e = idx & 15, lane = (idx >> 4) & 31, ks = (idx >> 9) & 7, nt = idx >> 12;
        int k = ks * 32 + (lane >> 4) * 8 + (e & 7) + (e >> 3) * 16, n = nt * 16 + (lane & 15);
        WxP[idx] = f2bf(Wx[k * N4 + n]);
    } else if (idx < 196608) {                // Wh_s [128,512]: 32 ntiles x 4 ksteps
        int j = idx - 131072;
        int e = j & 15, lane = (j >> 4) & 31, ks = (j >> 9) & 3, nt = j >> 11;
        int k = ks * 32 + (lane >> 4) * 8 + (e & 7) + (e >> 3) * 16, n = nt * 16 + (lane & 15);
        WhP[j] = f2bf(Wh[k * N4 + n]);
    } else {                                  // Wq/Wk/Wv/Wo [128,128]: 8 nt x 4 ks each
        int j = idx - 196608;
        int e = j & 15, lane = (j >> 4) & 31, ks = (j >> 9) & 3, nt = (j >> 11) & 7, mat = j >> 14;
        int k = ks * 32 + (lane >> 4) * 8 + (e & 7) + (e >> 3) * 16, n = nt * 16 + (lane & 15);
        const float* W = (mat == 0) ? Wq : (mat == 1) ? Wk : (mat == 2) ? Wv : Wo;
        WfP[j] = f2bf(W[k * HH + n]);
    }
}

// ---------------------------------------------------------------------------
// K1: PreX = x @ Wx_s + b_s, one block per step s, M=16 (batches), N=512, K=256.
// Output stored in D-fragment order: PreXp[((s*32+nt)*32+lane)*8 + i]
// ---------------------------------------------------------------------------
__global__ void k1_prex(const float* __restrict__ x, const float* __restrict__ bs,
                        const unsigned short* __restrict__ WxP, float* __restrict__ PreXp)
{
    int s = blockIdx.x;
    int t = threadIdx.x, w = t >> 5, lane = t & 31;
    int row = lane & 15, kh = (lane >> 4) * 8;
    // A fragments: row m = lane%16 of x[:, s, :], bf16-converted on the fly
    FragAB A[8];
    const float* xb = x + ((size_t)row * SS + s) * DD;
#pragma unroll
    for (int ks = 0; ks < 8; ++ks) {
        int k0 = ks * 32;
        v4f f0 = *(const v4f*)(xb + k0 + kh);
        v4f f1 = *(const v4f*)(xb + k0 + kh + 4);
        v4f f2 = *(const v4f*)(xb + k0 + 16 + kh);
        v4f f3 = *(const v4f*)(xb + k0 + 16 + kh + 4);
#pragma unroll
        for (int e = 0; e < 4; ++e) {
            A[ks].u[e]      = f2bf(f0[e]);
            A[ks].u[4 + e]  = f2bf(f1[e]);
            A[ks].u[8 + e]  = f2bf(f2[e]);
            A[ks].u[12 + e] = f2bf(f3[e]);
        }
    }
#pragma unroll
    for (int ntl = 0; ntl < 4; ++ntl) {
        int ntg = w * 4 + ntl;
        float bias = bs[ntg * 16 + row];
        Acc8 acc;
#pragma unroll
        for (int i = 0; i < 8; ++i) acc.f[i] = bias;
#pragma unroll
        for (int ks = 0; ks < 8; ++ks) {
            FragAB Bf;
            const unsigned short* p = WxP + ((size_t)(ntg * 8 + ks) * 32 + lane) * 16;
            Bf.i[0] = *(const v4i*)p;
            Bf.i[1] = *(const v4i*)(p + 8);
            acc.v = __builtin_amdgcn_wmma_f32_16x16x32_bf16(false, A[ks].v, false, Bf.v,
                                                            (short)0, acc.v, false, false);
        }
        float* op = PreXp + ((size_t)(s * 32 + ntg) * 32 + lane) * 8;
        *(v4f*)op       = acc.h[0];
        *(v4f*)(op + 4) = acc.h[1];
    }
}

// ---------------------------------------------------------------------------
// K2: sequential sLSTM scan. ONE workgroup (8 wave32). Wh_s fragments held in
// VGPRs for the whole scan (16 frags x 8 VGPRs per wave); PreX D-fragments
// double-buffered in registers so step s+1's global loads hide under step s.
// Per step: 128 x v_wmma_f32_16x16x32_bf16, gates, LayerNorm, emit HI (bf16).
// ---------------------------------------------------------------------------
#define PST 516        // preL row stride (floats)
#define AST 136        // actA row stride (bf16), padded against bank conflicts

__global__ __launch_bounds__(256, 1) void k2_scan(
        const unsigned short* __restrict__ WhP, const float* __restrict__ PreXp,
        const float* __restrict__ ln_g, const float* __restrict__ ln_b,
        unsigned short* __restrict__ HIg)
{
    __shared__ float          preL[BB * PST];   // 33 KB
    __shared__ unsigned short actA[BB * AST];   // 4.25 KB

    int t = threadIdx.x, w = t >> 5, lane = t & 31;
    int row = lane & 15, kh = (lane >> 4) * 8;

    // Wh_s B-fragments: loop-invariant, resident in VGPRs (128 VGPRs/lane)
    FragAB Bw[4][4];
#pragma unroll
    for (int ntl = 0; ntl < 4; ++ntl)
#pragma unroll
        for (int ks = 0; ks < 4; ++ks) {
            const unsigned short* p = WhP + (((w * 4 + ntl) * 4 + ks) * 32 + lane) * 16;
            Bw[ntl][ks].i[0] = *(const v4i*)p;
            Bw[ntl][ks].i[1] = *(const v4i*)(p + 8);
        }

    // h_s init = 0
    for (int i = t * 8; i < BB * AST; i += 2048)
        *(v4i*)(actA + i) = (v4i){0, 0, 0, 0};

    float c_reg[8], n_reg[8], gg[8], gb[8];
    int bEl = t >> 4, hl = t & 15;
#pragma unroll
    for (int j = 0; j < 8; ++j) {
        int h = hl + 16 * j;
        c_reg[j] = 0.0f; n_reg[j] = 1.0f;
        gg[j] = ln_g[h]; gb[j] = ln_b[h];
    }

    // preload PreX tiles for s = 0
    v4f curLo[4], curHi[4];
#pragma unroll
    for (int ntl = 0; ntl < 4; ++ntl) {
        const float* pxp = PreXp + ((size_t)(w * 4 + ntl) * 32 + lane) * 8;
        curLo[ntl] = *(const v4f*)pxp;
        curHi[ntl] = *(const v4f*)(pxp + 4);
    }
    __syncthreads();

    for (int s = 0; s < SS; ++s) {
        // issue next step's PreX loads early (latency hidden under this step)
        v4f nxtLo[4], nxtHi[4];
        int sn = (s + 1 < SS) ? s + 1 : s;
#pragma unroll
        for (int ntl = 0; ntl < 4; ++ntl) {
            const float* pxp = PreXp + ((size_t)(sn * 32 + w * 4 + ntl) * 32 + lane) * 8;
            nxtLo[ntl] = *(const v4f*)pxp;
            nxtHi[ntl] = *(const v4f*)(pxp + 4);
        }
        // ---- pre = PreX[:,s,:] + h_s @ Wh_s (WMMA) ----
        FragAB A[4];
#pragma unroll
        for (int ks = 0; ks < 4; ++ks) {
            const unsigned short* ap = actA + row * AST + ks * 32;
            A[ks].i[0] = *(const v4i*)(ap + kh);
            A[ks].i[1] = *(const v4i*)(ap + 16 + kh);
        }
#pragma unroll
        for (int ntl = 0; ntl < 4; ++ntl) {
            int ntg = w * 4 + ntl;
            Acc8 acc;
            acc.h[0] = curLo[ntl];
            acc.h[1] = curHi[ntl];
#pragma unroll
            for (int ks = 0; ks < 4; ++ks)
                acc.v = __builtin_amdgcn_wmma_f32_16x16x32_bf16(false, A[ks].v, false,
                        Bw[ntl][ks].v, (short)0, acc.v, false, false);
#pragma unroll
            for (int i = 0; i < 8; ++i)
                preL[(i + 8 * (lane >> 4)) * PST + ntg * 16 + row] = acc.f[i];
        }
        __syncthreads();

        // ---- gates + state update + LayerNorm (lane t: b=t/16, h=t%16+16j) ----
        float sum = 0.0f, ssq = 0.0f, hsv[8];
        const float* pr = preL + bEl * PST;
#pragma unroll
        for (int j = 0; j < 8; ++j) {
            int h = hl + 16 * j;
            float z  = fast_tanh(pr[h]);
            float ig = __expf(pr[128 + h]);
            float fg = sigm(pr[256 + h]);
            float og = sigm(pr[384 + h]);
            float c  = fg * c_reg[j] + ig * z;
            float n  = fg * n_reg[j] + ig;
            float hs = og * c / n;
            c_reg[j] = c; n_reg[j] = n; hsv[j] = hs;
            sum += hs; ssq += hs * hs;
        }
#pragma unroll
        for (int m = 1; m < 16; m <<= 1) {
            sum += __shfl_xor(sum, m, 16);
            ssq += __shfl_xor(ssq, m, 16);
        }
        float mu  = sum * (1.0f / 128.0f);
        float var = ssq * (1.0f / 128.0f) - mu * mu;
        float rs  = rsqrtf(var + 1e-5f);
#pragma unroll
        for (int j = 0; j < 8; ++j) {
            int h = hl + 16 * j;
            float hi = (hsv[j] - mu) * rs * gg[j] + gb[j];
            actA[bEl * AST + h] = f2bf(hsv[j]);                       // next-step A matrix
            HIg[((size_t)s * 16 + bEl) * 128 + h] = f2bf(hi);          // stream to mLSTM stage
        }
        __syncthreads();
#pragma unroll
        for (int ntl = 0; ntl < 4; ++ntl) { curLo[ntl] = nxtLo[ntl]; curHi[ntl] = nxtHi[ntl]; }
    }
}

// ---------------------------------------------------------------------------
// K3: Q/K/V/O = HI @ {Wq,Wk,Wv,Wo}. One block per step; wave -> (matrix, half).
// k scaled by 1/sqrt(H); sigmoid applied to O.
// ---------------------------------------------------------------------------
__global__ void k3_qkvo(const unsigned short* __restrict__ HIg, const unsigned short* __restrict__ WfP,
                        float* __restrict__ QT, float* __restrict__ KT,
                        float* __restrict__ VT, float* __restrict__ OT)
{
    int s = blockIdx.x;
    int t = threadIdx.x, w = t >> 5, lane = t & 31;
    int mat = w >> 1, half = w & 1;
    int row = lane & 15, kh = (lane >> 4) * 8;
    FragAB A[4];
    const unsigned short* hp = HIg + ((size_t)s * 16 + row) * 128;
#pragma unroll
    for (int ks = 0; ks < 4; ++ks) {
        A[ks].i[0] = *(const v4i*)(hp + ks * 32 + kh);
        A[ks].i[1] = *(const v4i*)(hp + ks * 32 + 16 + kh);
    }
    float* Out = (mat == 0) ? QT : (mat == 1) ? KT : (mat == 2) ? VT : OT;
#pragma unroll
    for (int ntl = 0; ntl < 4; ++ntl) {
        int nt = half * 4 + ntl;
        Acc8 acc;
#pragma unroll
        for (int i = 0; i < 8; ++i) acc.f[i] = 0.0f;
#pragma unroll
        for (int ks = 0; ks < 4; ++ks) {
            FragAB Bf;
            const unsigned short* p = WfP + (((size_t)(mat * 8 + nt) * 4 + ks) * 32 + lane) * 16;
            Bf.i[0] = *(const v4i*)p;
            Bf.i[1] = *(const v4i*)(p + 8);
            acc.v = __builtin_amdgcn_wmma_f32_16x16x32_bf16(false, A[ks].v, false, Bf.v,
                                                            (short)0, acc.v, false, false);
        }
#pragma unroll
        for (int i = 0; i < 8; ++i) {
            float v = acc.f[i];
            if (mat == 1) v *= 0.08838834764831845f;   // 1/sqrt(128)
            if (mat == 3) v = sigm(v);
            int b = i + 8 * (lane >> 4);
            Out[((size_t)s * 16 + b) * 128 + nt * 16 + row] = v;
        }
    }
}

// K3b: im = exp(hi.wi + bi), fm = sigmoid(hi.wf + bf), per (s,b).
__global__ void k3b_imfm(const unsigned short* __restrict__ HIg,
                         const float* __restrict__ wi, const float* __restrict__ bi,
                         const float* __restrict__ wf, const float* __restrict__ bfv,
                         float* __restrict__ IM, float* __restrict__ FM)
{
    int idx = blockIdx.x * 256 + threadIdx.x;      // = s*16 + b
    const v4i* hp = (const v4i*)(HIg + (size_t)idx * 128);
    float di = 0.0f, df = 0.0f;
#pragma unroll
    for (int c = 0; c < 16; ++c) {
        v4i chunk = hp[c];
#pragma unroll
        for (int q = 0; q < 4; ++q) {
            unsigned int u = (unsigned int)chunk[q];
            float lo = __uint_as_float(u << 16);
            float hi = __uint_as_float(u & 0xFFFF0000u);
            int h = c * 8 + q * 2;
            di += lo * wi[h] + hi * wi[h + 1];
            df += lo * wf[h] + hi * wf[h + 1];
        }
    }
    IM[idx] = __expf(di + bi[0]);
    FM[idx] = sigm(df + bfv[0]);
}

// ---------------------------------------------------------------------------
// K4: mLSTM matrix-memory scan. 64 blocks = 16 batches x 4 C-row groups.
// C slice [32][128] resident in LDS; n_m tracked redundantly per block.
// q/k/v/o/im/fm register double-buffered (s+1 loads issued during step s),
// global_prefetch for s+2. num = C_new q fused with the rank-1 update.
// ---------------------------------------------------------------------------
__global__ __launch_bounds__(256, 1) void k4_mlstm(
        const float* __restrict__ QT, const float* __restrict__ KT,
        const float* __restrict__ VT, const float* __restrict__ OT,
        const float* __restrict__ IM, const float* __restrict__ FM,
        float* __restrict__ out)
{
    __shared__ float Cl[32 * 132];
    __shared__ float q_s[128], k_s[128], v_s[32], o_s[32], nm[128], red[8];
    __shared__ float sIm, sFm;
    int t = threadIdx.x, w = t >> 5;
    int b = blockIdx.x >> 2, r0 = (blockIdx.x & 3) * 32;
    for (int i = t; i < 32 * 132; i += 256) Cl[i] = 0.0f;
    if (t < 128) nm[t] = 0.0f;
    int r = t >> 3, cb = (t & 7) * 16;

    // prolog: load step-0 staging into registers
    size_t base0 = (size_t)b * 128;
    float qkv = 0.0f, vov = 0.0f, imfm = 0.0f;
    if (t < 128) qkv = QT[base0 + t]; else qkv = KT[base0 + (t - 128)];
    if (t < 32) vov = VT[base0 + r0 + t]; else if (t < 64) vov = OT[base0 + r0 + (t - 32)];
    if (t == 64) imfm = IM[b]; else if (t == 65) imfm = FM[b];
    __syncthreads();

    for (int s = 0; s < SS; ++s) {
        // publish current staging to LDS
        if (t < 128) q_s[t] = qkv; else k_s[t - 128] = qkv;
        if (t < 32) v_s[t] = vov;
        else if (t < 64) o_s[t - 32] = vov;
        else if (t == 64) sIm = imfm;
        else if (t == 65) sFm = imfm;
        // issue next step's loads (complete while this step computes)
        float qkn = 0.0f, von = 0.0f, imfn = 0.0f;
        if (s + 1 < SS) {
            size_t nb = ((size_t)(s + 1) * 16 + b) * 128;
            if (t < 128) qkn = QT[nb + t]; else qkn = KT[nb + (t - 128)];
            if (t < 32) von = VT[nb + r0 + t]; else if (t < 64) von = OT[nb + r0 + (t - 32)];
            if (t == 64) imfn = IM[(s + 1) * 16 + b]; else if (t == 65) imfn = FM[(s + 1) * 16 + b];
        }
        if (s + 2 < SS && t < 16) {
            size_t pb = ((size_t)(s + 2) * 16 + b) * 128;
            __builtin_prefetch(&QT[pb + t * 8], 0, 1);
            __builtin_prefetch(&KT[pb + t * 8], 0, 1);
        }
        __syncthreads();

        float fm = sFm, im = sIm;
        // n_m update + den = max(|n_m . q|, 1)
        float p = 0.0f;
        if (t < 128) {
            float nv = fm * nm[t] + im * k_s[t];
            nm[t] = nv;
            p = nv * q_s[t];
        }
#pragma unroll
        for (int m = 1; m < 32; m <<= 1) p += __shfl_xor(p, m, 32);
        if ((t & 31) == 0) red[w] = p;
        __syncthreads();
        float den = fmaxf(fabsf(red[0] + red[1] + red[2] + red[3]), 1.0f);
        // fused C update + num = C_new . q  (thread: row r, 16-col strip cb)
        float ivr = im * v_s[r];
        float acc = 0.0f;
#pragma unroll
        for (int cc = 0; cc < 16; ++cc) {
            int c = cb + cc;
            float cn = fm * Cl[r * 132 + c] + ivr * k_s[c];
            Cl[r * 132 + c] = cn;
            acc += cn * q_s[c];
        }
#pragma unroll
        for (int m = 1; m < 8; m <<= 1) acc += __shfl_xor(acc, m, 8);
        if ((t & 7) == 0)
            out[((size_t)b * SS + s) * 128 + r0 + r] = o_s[r] * acc / den;
        __syncthreads();
        qkv = qkn; vov = von; imfm = imfn;
    }
}

// ---------------------------------------------------------------------------
extern "C" void kernel_launch(void* const* d_in, const int* in_sizes, int n_in,
                              void* d_out, int out_size, void* d_ws, size_t ws_size,
                              hipStream_t stream)
{
    const float* x   = (const float*)d_in[0];
    const float* Wx  = (const float*)d_in[1];
    const float* Wh  = (const float*)d_in[2];
    const float* bs  = (const float*)d_in[3];
    const float* lng = (const float*)d_in[4];
    const float* lnb = (const float*)d_in[5];
    const float* Wq  = (const float*)d_in[6];
    const float* Wk  = (const float*)d_in[7];
    const float* Wv  = (const float*)d_in[8];
    const float* Wo  = (const float*)d_in[9];
    const float* wi  = (const float*)d_in[10];
    const float* bi  = (const float*)d_in[11];
    const float* wf  = (const float*)d_in[12];
    const float* bf  = (const float*)d_in[13];

    char* ws = (char*)d_ws;
    unsigned short* WxP = (unsigned short*)(ws);                    // 256 KB
    unsigned short* WhP = (unsigned short*)(ws + (256 << 10));      // 128 KB
    unsigned short* WfP = (unsigned short*)(ws + (384 << 10));      // 128 KB
    float* PreXp        = (float*)(ws + (512 << 10));               // 32 MB
    char* p2            = ws + (512 << 10) + ((size_t)32 << 20);
    unsigned short* HIg = (unsigned short*)p2;                      // 4 MB
    float* QT = (float*)(p2 + ((size_t)4  << 20));                  // 8 MB each
    float* KT = (float*)(p2 + ((size_t)12 << 20));
    float* VT = (float*)(p2 + ((size_t)20 << 20));
    float* OT = (float*)(p2 + ((size_t)28 << 20));
    float* IM = (float*)(p2 + ((size_t)36 << 20));
    float* FM = (float*)(p2 + ((size_t)36 << 20) + (64 << 10));

    hipLaunchKernelGGL(k0_pack,  dim3(1024), dim3(256), 0, stream,
                       Wx, Wh, Wq, Wk, Wv, Wo, WxP, WhP, WfP);
    hipLaunchKernelGGL(k1_prex,  dim3(1024), dim3(256), 0, stream, x, bs, WxP, PreXp);
    hipLaunchKernelGGL(k2_scan,  dim3(1),    dim3(256), 0, stream,
                       WhP, PreXp, lng, lnb, HIg);
    hipLaunchKernelGGL(k3_qkvo,  dim3(1024), dim3(256), 0, stream, HIg, WfP, QT, KT, VT, OT);
    hipLaunchKernelGGL(k3b_imfm, dim3(64),   dim3(256), 0, stream, HIg, wi, bi, wf, bf, IM, FM);
    hipLaunchKernelGGL(k4_mlstm, dim3(64),   dim3(256), 0, stream,
                       QT, KT, VT, OT, IM, FM, (float*)d_out);
}
